// GraphAttnModel_32976758898728
// MI455X (gfx1250) — matrix operand order repaired and checked by reading, not compiled, not run on hardware
//
#include <hip/hip_runtime.h>

// ---------------------------------------------------------------------------
// GraphAttnModel for MI455X (gfx1250): bf16 WMMA GEMMs + wave32 edge softmax
// GEMM: 64x32 tile per wave, compile-time K, double-buffered fragments.
// ---------------------------------------------------------------------------

#define NND   20000   // nodes
#define FFD   142     // feature dim
#define HDD   128     // hidden dim (4 heads x 32)
#define KP0   160     // 142 padded to multiple of 32
#define KP1   128

typedef __attribute__((ext_vector_type(16))) __bf16        v16bf;
typedef __attribute__((ext_vector_type(8)))  float         v8f;
typedef __attribute__((ext_vector_type(8)))  unsigned int  v8u;

__device__ __forceinline__ unsigned short f2bf(float f) {
  unsigned int u = __float_as_uint(f);
  u += 0x7FFFu + ((u >> 16) & 1u);      // round-to-nearest-even
  return (unsigned short)(u >> 16);
}

// ---- pack weight f32 [K,Nc] row-major -> bf16 [Np,Kp] (transposed, padded) ----
__global__ void pack_w_kernel(const float* __restrict__ W, unsigned short* __restrict__ Bt,
                              int K, int Nc, int Kp, int Np) {
  int idx = blockIdx.x * blockDim.x + threadIdx.x;
  int total = Np * Kp;
  if (idx >= total) return;
  int n = idx / Kp, kk = idx - n * Kp;
  float v = (kk < K && n < Nc) ? W[(size_t)kk * Nc + n] : 0.f;
  Bt[idx] = f2bf(v);
}

// ---- pack activation f32 [M,K] -> bf16 [M,Kp] zero padded ----
__global__ void pack_a_kernel(const float* __restrict__ A, unsigned short* __restrict__ Ap,
                              int M, int K, int Kp) {
  int idx = blockIdx.x * blockDim.x + threadIdx.x;
  if (idx >= M * Kp) return;
  int m = idx / Kp, kk = idx - m * Kp;
  Ap[idx] = (kk < K) ? f2bf(A[(size_t)m * K + kk]) : (unsigned short)0;
}

// ---- LE2[c,n] = label_emb[c,:] @ W2[:,n] + b1[n] + b2[n]  (3 x 128, tiny) ----
__global__ void le2_kernel(const float* __restrict__ lemb, const float* __restrict__ W2,
                           const float* __restrict__ b1, const float* __restrict__ b2,
                           float* __restrict__ LE2) {
  int idx = blockIdx.x * blockDim.x + threadIdx.x;
  if (idx >= 3 * HDD) return;
  int c = idx >> 7, n = idx & 127;
  float s = b1[n] + b2[n];
  for (int kk = 0; kk < FFD; ++kk) s += lemb[c * FFD + kk] * W2[kk * HDD + n];
  LE2[idx] = s;
}

// ---- bf16 WMMA GEMM: C[M,Nc] = A[M,KP] * Bt[Np,KP]^T (+bias)(+addC) ----
// one wave per 64(M) x 32(N) tile; fully unrolled K with double-buffered frags
template <int KP>
__global__ __launch_bounds__(32) void gemm_wmma_bf16(
    const unsigned short* __restrict__ A, const unsigned short* __restrict__ Bt,
    const float* __restrict__ bias, const float* __restrict__ addC,
    float* __restrict__ C, int M, int Nc, int ldc) {
  constexpr int NS = KP / 32;
  const int lane = threadIdx.x & 31;
  const int ml = lane & 15;        // M (A) or N (B/C) index within 16-tile
  const int kh = lane >> 4;        // K-half selector
  const int m_base = blockIdx.x * 64;
  const int n0 = blockIdx.y * 32;

  const unsigned short* brow[2];
  brow[0] = Bt + (size_t)(n0 + ml) * KP;
  brow[1] = Bt + (size_t)(n0 + 16 + ml) * KP;
  const unsigned short* arow[4];
#pragma unroll
  for (int t = 0; t < 4; ++t) {
    int m = m_base + t * 16 + ml;
    if (m >= M) m = M - 1;          // clamp loads; stores are guarded
    arow[t] = A + (size_t)m * KP;
  }

  v8f acc[4][2] = {};
  v8u bfr[2][2];                    // [buffer][n-subtile]
  v8u afr[2][4];                    // [buffer][m-subtile]

  auto load_step = [&](int s, int buf) {
    const int ks = s * 32;
#pragma unroll
    for (int j = 0; j < 8; ++j) {
      bfr[buf][0][j] = *(const unsigned int*)(brow[0] + ks + kh * 16 + 2 * j);
      bfr[buf][1][j] = *(const unsigned int*)(brow[1] + ks + kh * 16 + 2 * j);
    }
#pragma unroll
    for (int t = 0; t < 4; ++t) {
#pragma unroll
      for (int j = 0; j < 4; ++j) {
        afr[buf][t][j]     = *(const unsigned int*)(arow[t] + ks + kh * 16 + 2 * j);
        afr[buf][t][4 + j] = *(const unsigned int*)(arow[t] + ks + kh * 16 + 8 + 2 * j);
      }
    }
  };

  load_step(0, 0);
#pragma unroll
  for (int s = 0; s < NS; ++s) {
    const int cur = s & 1, nxt = cur ^ 1;
    if (s + 1 < NS) load_step(s + 1, nxt);   // prefetch next K-step
#pragma unroll
    for (int t = 0; t < 4; ++t) {
#pragma unroll
      for (int c = 0; c < 2; ++c) {
        acc[t][c] = __builtin_amdgcn_wmma_f32_16x16x32_bf16(
            false, __builtin_bit_cast(v16bf, afr[cur][t]),
            false, __builtin_bit_cast(v16bf, bfr[cur][c]),
            (short)0, acc[t][c], false, false);
      }
    }
  }

#pragma unroll
  for (int c = 0; c < 2; ++c) {
    const int col = n0 + c * 16 + ml;
    if (col < Nc) {
      float bv = bias ? bias[col] : 0.f;
#pragma unroll
      for (int t = 0; t < 4; ++t) {
#pragma unroll
        for (int r = 0; r < 8; ++r) {
          int m = m_base + t * 16 + kh * 8 + r;
          if (m < M) {
            float vv = acc[t][c][r] + bv;
            if (addC) vv += addC[(size_t)m * ldc + col];
            C[(size_t)m * ldc + col] = vv;
          }
        }
      }
    }
  }
}

// ---- t1b = bf16( prelu( (t0 + LE2[label]) * BN*g + b , a3) )  [N,128] ----
__global__ void fuse0_kernel(const float* __restrict__ t0, const float* __restrict__ LE2,
                             const int* __restrict__ labels, const float* __restrict__ g,
                             const float* __restrict__ b, const float* __restrict__ aP,
                             unsigned short* __restrict__ out, int n) {
  int idx = blockIdx.x * blockDim.x + threadIdx.x;
  if (idx >= n * HDD) return;
  int i = idx >> 7, c = idx & 127;
  float v = t0[idx] + LE2[labels[i] * HDD + c];
  v = v * (0.99999500f * g[c]) + b[c];               // BN: 1/sqrt(1+1e-5)
  float a = aP[0];
  v = (v >= 0.f) ? v : a * v;
  out[idx] = f2bf(v);
}

// ---- x1b = bf16( prelu( x*BN*g + b , a ) ) ----
__global__ void bnact_kernel(const float* __restrict__ x, const float* __restrict__ g,
                             const float* __restrict__ b, const float* __restrict__ aP,
                             unsigned short* __restrict__ out, int n) {
  int idx = blockIdx.x * blockDim.x + threadIdx.x;
  if (idx >= n * HDD) return;
  int c = idx & 127;
  float v = x[idx] * (0.99999500f * g[c]) + b[c];
  float a = aP[0];
  v = (v >= 0.f) ? v : a * v;
  out[idx] = f2bf(v);
}

// ---- attention: one wave per (node, head); 32 lanes <-> 32 incoming edges ----
// edge_dst == repeat(arange(N),32)  =>  edges of node i are [i*32,(i+1)*32)
__global__ __launch_bounds__(256) void attn_kernel(
    const float* __restrict__ q, const float* __restrict__ k, const float* __restrict__ v,
    const int* __restrict__ esrc, float* __restrict__ agg, int n_nodes) {
  int wid = (blockIdx.x * blockDim.x + threadIdx.x) >> 5;
  int lane = threadIdx.x & 31;
  int node = wid >> 2;
  int head = wid & 3;
  if (node >= n_nodes) return;
  int src = esrc[node * 32 + lane];
  const float* qr = q + (size_t)src * HDD + head * 32;
  const float* kr = k + (size_t)node * HDD + head * 32;
  float s = 0.f;
#pragma unroll
  for (int d = 0; d < 32; d += 4) {
    float4 qa = *(const float4*)(qr + d);
    float4 ka = *(const float4*)(kr + d);
    s += qa.x * ka.x + qa.y * ka.y + qa.z * ka.z + qa.w * ka.w;
  }
  s *= 0.17677669529f;                               // 1/sqrt(32)
  float mx = s;
#pragma unroll
  for (int off = 16; off >= 1; off >>= 1) mx = fmaxf(mx, __shfl_xor(mx, off, 32));
  float p = __expf(s - mx);
  float sum = p;
#pragma unroll
  for (int off = 16; off >= 1; off >>= 1) sum += __shfl_xor(sum, off, 32);
  float w = p / sum;
  // lane now = output dim; broadcast each edge's weight/src and accumulate
  float acc = 0.f;
#pragma unroll
  for (int e = 0; e < 32; ++e) {
    float we = __shfl(w, e, 32);
    int   se = __shfl(src, e, 32);
    acc += we * v[(size_t)se * HDD + head * 32 + lane];
  }
  agg[(size_t)node * HDD + head * 32 + lane] = acc;
}

// ---- gate + LayerNorm + PReLU, one wave per node; output bf16 [N,128] ----
__global__ __launch_bounds__(256) void gate_ln_kernel(
    const float* __restrict__ skip, const float* __restrict__ agg,
    const float* __restrict__ wg, const float* __restrict__ bg,
    const float* __restrict__ lng, const float* __restrict__ lnb,
    const float* __restrict__ aP, unsigned short* __restrict__ outb, int n_nodes) {
  int wid = (blockIdx.x * blockDim.x + threadIdx.x) >> 5;
  int lane = threadIdx.x & 31;
  if (wid >= n_nodes) return;
  const float* sp = skip + (size_t)wid * HDD;
  const float* ap = agg + (size_t)wid * HDD;
  float sk[4], ag[4];
  float dot = 0.f;
#pragma unroll
  for (int j = 0; j < 4; ++j) {
    int d = lane + 32 * j;
    sk[j] = sp[d]; ag[j] = ap[d];
    dot += sk[j] * wg[d] + ag[j] * wg[HDD + d] + (sk[j] - ag[j]) * wg[2 * HDD + d];
  }
#pragma unroll
  for (int off = 16; off >= 1; off >>= 1) dot += __shfl_xor(dot, off, 32);
  float g = 1.f / (1.f + __expf(-(dot + bg[0])));
  float r[4], s1 = 0.f, s2 = 0.f;
#pragma unroll
  for (int j = 0; j < 4; ++j) {
    r[j] = g * sk[j] + (1.f - g) * ag[j];
    s1 += r[j]; s2 += r[j] * r[j];
  }
#pragma unroll
  for (int off = 16; off >= 1; off >>= 1) { s1 += __shfl_xor(s1, off, 32); s2 += __shfl_xor(s2, off, 32); }
  float mean = s1 * (1.f / 128.f);
  float var  = s2 * (1.f / 128.f) - mean * mean;
  float inv  = rsqrtf(var + 1e-5f);
  float a = aP[0];
#pragma unroll
  for (int j = 0; j < 4; ++j) {
    int d = lane + 32 * j;
    float y = (r[j] - mean) * inv * lng[d] + lnb[d];
    y = (y >= 0.f) ? y : a * y;
    outb[(size_t)wid * HDD + d] = f2bf(y);
  }
}

// ---------------------------------------------------------------------------
extern "C" void kernel_launch(void* const* d_in, const int* in_sizes, int n_in,
                              void* d_out, int out_size, void* d_ws, size_t ws_size,
                              hipStream_t stream) {
  (void)in_sizes; (void)n_in; (void)out_size; (void)ws_size;
  const int N = NND, F = FFD, HD = HDD;

  const float* features = (const float*)d_in[0];
  const float* lemb     = (const float*)d_in[1];
  const float* W1 = (const float*)d_in[2];  const float* b1 = (const float*)d_in[3];
  const float* W2 = (const float*)d_in[4];  const float* b2 = (const float*)d_in[5];
  const float* bn0g = (const float*)d_in[6]; const float* bn0b = (const float*)d_in[7];
  const float* a3 = (const float*)d_in[8];
  const float* W3 = (const float*)d_in[9];  const float* b3 = (const float*)d_in[10];
  const float* Wq[2] = {(const float*)d_in[11], (const float*)d_in[23]};
  const float* bq[2] = {(const float*)d_in[12], (const float*)d_in[24]};
  const float* Wk[2] = {(const float*)d_in[13], (const float*)d_in[25]};
  const float* bk[2] = {(const float*)d_in[14], (const float*)d_in[26]};
  const float* Wv[2] = {(const float*)d_in[15], (const float*)d_in[27]};
  const float* bv[2] = {(const float*)d_in[16], (const float*)d_in[28]};
  const float* Ws[2] = {(const float*)d_in[17], (const float*)d_in[29]};
  const float* bs[2] = {(const float*)d_in[18], (const float*)d_in[30]};
  const float* Wg[2] = {(const float*)d_in[19], (const float*)d_in[31]};
  const float* bg[2] = {(const float*)d_in[20], (const float*)d_in[32]};
  const float* lng[2] = {(const float*)d_in[21], (const float*)d_in[33]};
  const float* lnb[2] = {(const float*)d_in[22], (const float*)d_in[34]};
  const float* aconv = (const float*)d_in[35];
  const float* W4 = (const float*)d_in[36]; const float* b4 = (const float*)d_in[37];
  const float* bn1g = (const float*)d_in[38]; const float* bn1b = (const float*)d_in[39];
  const float* a5 = (const float*)d_in[40];
  const float* W5 = (const float*)d_in[41]; const float* b5 = (const float*)d_in[42];
  const int* labels = (const int*)d_in[43];
  const int* esrc   = (const int*)d_in[44];
  float* out = (float*)d_out;

  // ---- workspace carve-out (256B aligned) ----
  char* base = (char*)d_ws;
  size_t off = 0;
  auto alloc = [&](size_t bytes) { size_t o = off; off = (off + bytes + 255) & ~(size_t)255; return base + o; };
  unsigned short* bufA = (unsigned short*)alloc((size_t)N * KP0 * 2);  // featA -> hA -> x1b
  float*          bufT = (float*)alloc((size_t)N * HD * 4);           // t0 -> x1
  unsigned short* bufH = (unsigned short*)alloc((size_t)N * HD * 2);  // t1b -> hb (layer io)
  float*          bufF = (float*)alloc((size_t)N * F * 4);            // hf -> agg
  float* qb   = (float*)alloc((size_t)N * HD * 4);
  float* kb   = (float*)alloc((size_t)N * HD * 4);
  float* vb   = (float*)alloc((size_t)N * HD * 4);
  float* skb  = (float*)alloc((size_t)N * HD * 4);
  float* LE2  = (float*)alloc(3 * HD * 4);
  unsigned short* W1t = (unsigned short*)alloc((size_t)HD * KP0 * 2);
  unsigned short* W3t = (unsigned short*)alloc((size_t)160 * KP1 * 2); // 142 -> 160 rows
  unsigned short* Wqt[2], *Wkt[2], *Wvt[2], *Wst[2];
  for (int l = 0; l < 2; ++l) {
    int kp = l ? KP1 : KP0;
    Wqt[l] = (unsigned short*)alloc((size_t)HD * kp * 2);
    Wkt[l] = (unsigned short*)alloc((size_t)HD * kp * 2);
    Wvt[l] = (unsigned short*)alloc((size_t)HD * kp * 2);
    Wst[l] = (unsigned short*)alloc((size_t)HD * kp * 2);
  }
  unsigned short* W4t = (unsigned short*)alloc((size_t)HD * KP1 * 2);
  unsigned short* W5t = (unsigned short*)alloc((size_t)32 * KP1 * 2);  // 2 -> 32 rows

  auto packw = [&](const float* W, unsigned short* Bt, int K, int Nc, int Kp, int Np) {
    int total = Np * Kp;
    pack_w_kernel<<<(total + 255) / 256, 256, 0, stream>>>(W, Bt, K, Nc, Kp, Np);
  };
  auto gemm = [&](const unsigned short* A, const unsigned short* Bt, const float* bias,
                  const float* addC, float* C, int M, int Nc, int Kp) {
    dim3 grd((M + 63) / 64, (Nc + 31) / 32);
    if (Kp == KP0)
      gemm_wmma_bf16<KP0><<<grd, 32, 0, stream>>>(A, Bt, bias, addC, C, M, Nc, Nc);
    else
      gemm_wmma_bf16<KP1><<<grd, 32, 0, stream>>>(A, Bt, bias, addC, C, M, Nc, Nc);
  };

  // ---- stage 0: pack all weights ----
  packw(W1, W1t, F, HD, KP0, HD);
  packw(W3, W3t, HD, F, KP1, 160);
  for (int l = 0; l < 2; ++l) {
    int kin = l ? HD : F, kp = l ? KP1 : KP0;
    packw(Wq[l], Wqt[l], kin, HD, kp, HD);
    packw(Wk[l], Wkt[l], kin, HD, kp, HD);
    packw(Wv[l], Wvt[l], kin, HD, kp, HD);
    packw(Ws[l], Wst[l], kin, HD, kp, HD);
  }
  packw(W4, W4t, HD, HD, KP1, HD);
  packw(W5, W5t, HD, 2, KP1, 32);
  le2_kernel<<<2, 256, 0, stream>>>(lemb, W2, b1, b2, LE2);

  // ---- stage 1: t0 = feat @ W1 ; t1b = pack(prelu(BN(t0 + LE2[label]))) ----
  pack_a_kernel<<<((N * KP0) + 255) / 256, 256, 0, stream>>>(features, bufA, N, F, KP0);
  gemm(bufA, W1t, nullptr, nullptr, bufT, N, HD, KP0);
  fuse0_kernel<<<((N * HD) + 255) / 256, 256, 0, stream>>>(bufT, LE2, labels, bn0g, bn0b, a3, bufH, N);

  // ---- stage 2: h = features + t1b @ W3 + b3 ; pack to hA ----
  gemm(bufH, W3t, b3, features, bufF, N, F, KP1);
  pack_a_kernel<<<((N * KP0) + 255) / 256, 256, 0, stream>>>(bufF, bufA, N, F, KP0);

  // ---- conv layers ----
  const int attn_grid = (N * 4 * 32 + 255) / 256;
  const int node_grid = (N * 32 + 255) / 256;
  for (int l = 0; l < 2; ++l) {
    const unsigned short* hin = l ? bufH : bufA;
    int kp = l ? KP1 : KP0;
    gemm(hin, Wqt[l], bq[l], nullptr, qb,  N, HD, kp);
    gemm(hin, Wkt[l], bk[l], nullptr, kb,  N, HD, kp);
    gemm(hin, Wvt[l], bv[l], nullptr, vb,  N, HD, kp);
    gemm(hin, Wst[l], bs[l], nullptr, skb, N, HD, kp);
    attn_kernel<<<attn_grid, 256, 0, stream>>>(qb, kb, vb, esrc, bufF, N);
    gate_ln_kernel<<<node_grid, 256, 0, stream>>>(skb, bufF, Wg[l], bg[l], lng[l], lnb[l],
                                                  aconv, bufH, N);
  }

  // ---- head: x = h @ W4 + b4 ; prelu(BN(x)) ; out = x @ W5 + b5 ----
  gemm(bufH, W4t, b4, nullptr, bufT, N, HD, KP1);
  bnact_kernel<<<((N * HD) + 255) / 256, 256, 0, stream>>>(bufT, bn1g, bn1b, a5, bufA /*as x1b*/, N);
  gemm((unsigned short*)bufA, W5t, b5, nullptr, out, N, 2, KP1);
}
